// MultiHeadAttention_48902497632771
// MI455X (gfx1250) — matrix operand order repaired and checked by reading, not compiled, not run on hardware
//
#include <hip/hip_runtime.h>
#include <cstdint>
#include <cmath>

typedef __attribute__((ext_vector_type(16))) __bf16 v16bf;
typedef __attribute__((ext_vector_type(8)))  float  v8f;
typedef uint32_t u32x4 __attribute__((ext_vector_type(4)));
typedef int      i32x4 __attribute__((ext_vector_type(4)));
typedef int      i32x8 __attribute__((ext_vector_type(8)));

#if defined(__gfx1250__) && __has_builtin(__builtin_amdgcn_tensor_load_to_lds)
#define USE_TDM 1
#else
#define USE_TDM 0
#endif

union BFrag {
  v16bf v;
  uint4 q[2];
};

__device__ inline uint16_t f2bf(float f) {
  uint32_t u = __float_as_uint(f);
  u = (u + 0x7FFFu + ((u >> 16) & 1u)) >> 16;   // round-to-nearest-even
  return (uint16_t)u;
}
__device__ inline float bf2f(uint16_t h) {
  return __uint_as_float(((uint32_t)h) << 16);
}
__device__ inline v8f wmma_bf16(v16bf a, v16bf b, v8f c) {
  return __builtin_amdgcn_wmma_f32_16x16x32_bf16(false, a, false, b, (short)0, c,
                                                 false, false);
}

// ds_swizzle group-of-32 XOR butterfly: offset = xor_mask<<10 | and_mask(0x1f).
// xor masks 1/2/4/8 stay inside each 16-lane half, preserving per-row reductions.
#define SWZ_XOR_F(x, m) \
  __int_as_float(__builtin_amdgcn_ds_swizzle(__float_as_int(x), ((m) << 10) | 31))

// ---------------------------------------------------------------------------
// GEMM: C[M,N] = A[M,K] * W[N,K]^T   (y = A @ W.T, W row-major [N][K])
// A f32 (converted to bf16 while staging) or bf16; C bf16 or f32.
// Block: 256 threads (8 waves), tile 128x128, K-step 32, double-buffered LDS,
// global_prefetch two K-steps ahead.
// ---------------------------------------------------------------------------
template <bool A_F32, bool OUT_F32>
__global__ __launch_bounds__(256) void gemm_bf16_wmma(
    const void* __restrict__ Ap, const float* __restrict__ W,
    void* __restrict__ Cp, int M, int N, int K) {
  __shared__ __align__(16) uint16_t As[2][128 * 32];
  __shared__ __align__(16) uint16_t Bs[2][128 * 32];

  const int tid = threadIdx.x;
  const int wave = tid >> 5, lane = tid & 31;
  const int half = lane >> 4, ln = lane & 15;
  const int wr = wave >> 2, wc = wave & 3;   // 2x4 wave grid: 64x32 per wave
  const int Mb = blockIdx.x * 128, Nb = blockIdx.y * 128;

  const v8f vzero = {0, 0, 0, 0, 0, 0, 0, 0};
  v8f acc[4][2];
#pragma unroll
  for (int mi = 0; mi < 4; ++mi)
#pragma unroll
    for (int ni = 0; ni < 2; ++ni) acc[mi][ni] = vzero;

  auto stage = [&](int kt, int buf) {
    const int k0 = kt * 32;
    const bool pf = (k0 + 64 + 32) <= K;   // prefetch 2 steps ahead
    if constexpr (A_F32) {
      const float* A = (const float*)Ap;
#pragma unroll
      for (int i = 0; i < 4; ++i) {
        const int idx = tid + i * 256;
        const int row = idx >> 3;
        const int c4 = (idx & 7) * 4;
        const float* src = A + (size_t)(Mb + row) * K + k0 + c4;
        if (i == 0 && pf) __builtin_prefetch(src + 64, 0, 3);
        float4 f = *(const float4*)src;
        uint32_t lo = (uint32_t)f2bf(f.x) | ((uint32_t)f2bf(f.y) << 16);
        uint32_t hi = (uint32_t)f2bf(f.z) | ((uint32_t)f2bf(f.w) << 16);
        *(uint2*)&As[buf][row * 32 + c4] = make_uint2(lo, hi);
      }
    } else {
      const uint16_t* A = (const uint16_t*)Ap;
#pragma unroll
      for (int i = 0; i < 2; ++i) {
        const int idx = tid + i * 256;
        const int row = idx >> 2;
        const int c8 = (idx & 3) * 8;
        const uint16_t* src = A + (size_t)(Mb + row) * K + k0 + c8;
        if (i == 0 && pf) __builtin_prefetch(src + 64, 0, 3);
        *(uint4*)&As[buf][row * 32 + c8] = *(const uint4*)src;
      }
    }
#pragma unroll
    for (int i = 0; i < 4; ++i) {
      const int idx = tid + i * 256;
      const int row = idx >> 3;
      const int c4 = (idx & 7) * 4;
      const float* src = W + (size_t)(Nb + row) * K + k0 + c4;
      if (i == 0 && pf) __builtin_prefetch(src + 64, 0, 3);
      float4 f = *(const float4*)src;
      uint32_t lo = (uint32_t)f2bf(f.x) | ((uint32_t)f2bf(f.y) << 16);
      uint32_t hi = (uint32_t)f2bf(f.z) | ((uint32_t)f2bf(f.w) << 16);
      *(uint2*)&Bs[buf][row * 32 + c4] = make_uint2(lo, hi);
    }
  };

  const int nk = K >> 5;
  stage(0, 0);
  for (int kt = 0; kt < nk; ++kt) {
    __syncthreads();
    if (kt + 1 < nk) stage(kt + 1, (kt + 1) & 1);
    const int buf = kt & 1;

    BFrag a[4], b[2];
#pragma unroll
    for (int mi = 0; mi < 4; ++mi) {
      const int m = wr * 64 + mi * 16 + ln;
      a[mi].q[0] = *(const uint4*)&As[buf][m * 32 + half * 8];
      a[mi].q[1] = *(const uint4*)&As[buf][m * 32 + 16 + half * 8];
    }
#pragma unroll
    for (int ni = 0; ni < 2; ++ni) {
      const int n = wc * 32 + ni * 16 + ln;
      b[ni].q[0] = *(const uint4*)&Bs[buf][n * 32 + half * 8];
      b[ni].q[1] = *(const uint4*)&Bs[buf][n * 32 + 16 + half * 8];
    }
#pragma unroll
    for (int mi = 0; mi < 4; ++mi)
#pragma unroll
      for (int ni = 0; ni < 2; ++ni)
        acc[mi][ni] = wmma_bf16(a[mi].v, b[ni].v, acc[mi][ni]);
  }

#pragma unroll
  for (int mi = 0; mi < 4; ++mi)
#pragma unroll
    for (int ni = 0; ni < 2; ++ni)
#pragma unroll
      for (int r = 0; r < 8; ++r) {
        const int row = Mb + wr * 64 + mi * 16 + r + half * 8;
        const int col = Nb + wc * 32 + ni * 16 + ln;
        const float val = acc[mi][ni][r];
        if constexpr (OUT_F32)
          ((float*)Cp)[(size_t)row * N + col] = val;
        else
          ((uint16_t*)Cp)[(size_t)row * N + col] = f2bf(val);
      }
}

// ---------------------------------------------------------------------------
// RoPE + relayout: raw [B*T][H*64] bf16 -> head-major [B*H][T][64] bf16.
// ---------------------------------------------------------------------------
__global__ __launch_bounds__(256) void rope_pack(
    const uint16_t* __restrict__ qr, const uint16_t* __restrict__ kr,
    const uint16_t* __restrict__ vr, uint16_t* __restrict__ Qh,
    uint16_t* __restrict__ Kh, uint16_t* __restrict__ Vh) {
  const int idx = blockIdx.x * 256 + threadIdx.x;  // B*T*H*32 = 1<<22
  const int d = idx & 31;
  const int h = (idx >> 5) & 15;
  const int t = (idx >> 9) & 2047;
  const int b = idx >> 20;

  const size_t raw = ((size_t)b * 2048 + t) * 1024 + (size_t)h * 64;
  const size_t out = (((size_t)b * 16 + h) * 2048 + t) * 64;

  const float inv = __expf(-(float)d * (logf(10000.0f) / 32.0f));
  const float ang = (float)t * inv;
  float sn, cs;
  __sincosf(ang, &sn, &cs);

  const float q1 = bf2f(qr[raw + d]), q2 = bf2f(qr[raw + d + 32]);
  Qh[out + d] = f2bf(q1 * cs - q2 * sn);
  Qh[out + d + 32] = f2bf(q2 * cs + q1 * sn);

  const float k1 = bf2f(kr[raw + d]), k2 = bf2f(kr[raw + d + 32]);
  Kh[out + d] = f2bf(k1 * cs - k2 * sn);
  Kh[out + d + 32] = f2bf(k2 * cs + k1 * sn);

  Vh[out + d] = vr[raw + d];
  Vh[out + d + 32] = vr[raw + d + 32];
}

// ---------------------------------------------------------------------------
// Flash attention (causal). K-chunk staged by the Tensor Data Mover
// (tensor_load_to_lds, 2-D D#, wave0 issues, s_wait_tensorcnt before the
// barrier); V staged manually (needs d-major transpose). Row reductions use
// ds_swizzle XOR butterflies.
// ---------------------------------------------------------------------------
__global__ __launch_bounds__(256) void attn_flash(
    const uint16_t* __restrict__ Qh, const uint16_t* __restrict__ Kh,
    const uint16_t* __restrict__ Vh, uint16_t* __restrict__ Oh) {
  __shared__ __align__(16) uint16_t Kc[32 * 64];      // [key][d]
  __shared__ __align__(16) uint16_t Vt[64 * 32];      // [d][key]
  __shared__ __align__(16) uint16_t Pbuf[8 * 16 * 32];

  const int tid = threadIdx.x;
  const int wave = tid >> 5, lane = tid & 31;
  const int half = lane >> 4, ln = lane & 15;
  const int mtile = blockIdx.x;
  const int bh = blockIdx.y;
  const int tq0 = mtile * 128 + wave * 16;
  uint16_t* Pw = Pbuf + wave * (16 * 32);

  BFrag aQ[2];
  const uint16_t* Qrow = Qh + ((size_t)bh * 2048 + tq0 + ln) * 64;
#pragma unroll
  for (int f = 0; f < 2; ++f) {
    aQ[f].q[0] = *(const uint4*)(Qrow + f * 32 + half * 8);
    aQ[f].q[1] = *(const uint4*)(Qrow + f * 32 + 16 + half * 8);
  }

  float mrow[8], lrow[8];
#pragma unroll
  for (int r = 0; r < 8; ++r) {
    mrow[r] = -__builtin_inff();
    lrow[r] = 0.0f;
  }
  const v8f vzero = {0, 0, 0, 0, 0, 0, 0, 0};
  v8f o[4];
#pragma unroll
  for (int f = 0; f < 4; ++f) o[f] = vzero;

  const int nchunks = mtile * 4 + 4;
  const int mylast = (tq0 + 15) >> 5;
  const float scale = 0.125f;  // 1/sqrt(64)

  const int rowk = tid >> 3;
  const int c8 = (tid & 7) * 8;

  for (int ch = 0; ch < nchunks; ++ch) {
    const int sbase = ch * 32;
    const size_t gro = ((size_t)bh * 2048 + sbase + rowk) * 64 + c8;

    {  // V: cooperative transpose into LDS [d][key]
      uint4 vv = *(const uint4*)(Vh + gro);
      const uint16_t* pv = (const uint16_t*)&vv;
#pragma unroll
      for (int j = 0; j < 8; ++j) Vt[(c8 + j) * 32 + rowk] = pv[j];
    }

#if USE_TDM
    if (wave == 0) {
      // Tensor DMA: 32x64 bf16 row-major tile, global -> LDS (Kc).
      const uint64_t gk =
          (uint64_t)(uintptr_t)(Kh + ((size_t)bh * 2048 + sbase) * 64);
      const uint32_t lds = (uint32_t)(uintptr_t)(&Kc[0]);
      // D# group0: count=1 | lds_addr | global_addr | type=2 ("image")
      u32x4 g0 = {1u, lds, (uint32_t)gk,
                  (uint32_t)((gk >> 32) & 0x01FFFFFFu) | (2u << 30)};
      // D# group1: data_size=2B; tensor_dim0=64; tensor_dim1=2048;
      // tile_dim0=64; tile_dim1=32; tensor_dim0_stride=64.
      i32x8 g1 = {(int)(1u << 16), (int)(64u << 16), (int)(2048u << 16),
                  (int)(64u << 16), 32, 64, 0, 0};
      i32x4 gz = {0, 0, 0, 0};
#if __clang_major__ >= 23
      i32x8 gz8 = {0, 0, 0, 0, 0, 0, 0, 0};
      __builtin_amdgcn_tensor_load_to_lds(g0, g1, gz, gz, gz8, 0);
#else
      __builtin_amdgcn_tensor_load_to_lds(g0, g1, gz, gz, 0);
#endif
      __builtin_amdgcn_s_wait_tensorcnt(0);
    }
#else
    *(uint4*)&Kc[rowk * 64 + c8] = *(const uint4*)(Kh + gro);
#endif
    __syncthreads();

    if (ch <= mylast) {
      v8f s[2];
#pragma unroll
      for (int st = 0; st < 2; ++st) {
        const uint16_t* krp = &Kc[(st * 16 + ln) * 64];
        BFrag bk0, bk1;
        bk0.q[0] = *(const uint4*)(krp + half * 8);
        bk0.q[1] = *(const uint4*)(krp + 16 + half * 8);
        bk1.q[0] = *(const uint4*)(krp + 32 + half * 8);
        bk1.q[1] = *(const uint4*)(krp + 48 + half * 8);
        v8f t = wmma_bf16(aQ[0].v, bk0.v, vzero);
        s[st] = wmma_bf16(aQ[1].v, bk1.v, t);
      }

      float alpha[8];
#pragma unroll
      for (int r = 0; r < 8; ++r) {
        const int qrow = tq0 + r + half * 8;
        float v0 = s[0][r] * scale;
        float v1 = s[1][r] * scale;
        if (sbase + ln > qrow) v0 = -__builtin_inff();
        if (sbase + 16 + ln > qrow) v1 = -__builtin_inff();
        float mx = fmaxf(v0, v1);
        mx = fmaxf(mx, SWZ_XOR_F(mx, 8));
        mx = fmaxf(mx, SWZ_XOR_F(mx, 4));
        mx = fmaxf(mx, SWZ_XOR_F(mx, 2));
        mx = fmaxf(mx, SWZ_XOR_F(mx, 1));
        const float mn = fmaxf(mrow[r], mx);
        const float al = __expf(mrow[r] - mn);
        const float p0 = __expf(v0 - mn);
        const float p1 = __expf(v1 - mn);
        float sum = p0 + p1;
        sum += SWZ_XOR_F(sum, 8);
        sum += SWZ_XOR_F(sum, 4);
        sum += SWZ_XOR_F(sum, 2);
        sum += SWZ_XOR_F(sum, 1);
        lrow[r] = lrow[r] * al + sum;
        mrow[r] = mn;
        alpha[r] = al;
        Pw[(r + half * 8) * 32 + ln] = f2bf(p0);
        Pw[(r + half * 8) * 32 + 16 + ln] = f2bf(p1);
      }
#pragma unroll
      for (int f = 0; f < 4; ++f)
#pragma unroll
        for (int r = 0; r < 8; ++r) o[f][r] *= alpha[r];

      BFrag aP;
      aP.q[0] = *(const uint4*)&Pw[ln * 32 + half * 8];
      aP.q[1] = *(const uint4*)&Pw[ln * 32 + 16 + half * 8];

#pragma unroll
      for (int f = 0; f < 4; ++f) {
        const uint16_t* vrp = &Vt[(f * 16 + ln) * 32];
        BFrag bv;
        bv.q[0] = *(const uint4*)(vrp + half * 8);
        bv.q[1] = *(const uint4*)(vrp + 16 + half * 8);
        o[f] = wmma_bf16(aP.v, bv.v, o[f]);
      }
    }
    __syncthreads();
  }

  float invl[8];
#pragma unroll
  for (int r = 0; r < 8; ++r) invl[r] = 1.0f / lrow[r];
  const int b = bh >> 4, h = bh & 15;
#pragma unroll
  for (int f = 0; f < 4; ++f)
#pragma unroll
    for (int r = 0; r < 8; ++r) {
      const size_t row = (size_t)b * 2048 + tq0 + r + half * 8;
      Oh[row * 1024 + h * 64 + f * 16 + ln] = f2bf(o[f][r] * invl[r]);
    }
}

// ---------------------------------------------------------------------------
// Host-side orchestration.
// ---------------------------------------------------------------------------
extern "C" void kernel_launch(void* const* d_in, const int* in_sizes, int n_in,
                              void* d_out, int out_size, void* d_ws,
                              size_t ws_size, hipStream_t stream) {
  (void)in_sizes; (void)n_in; (void)out_size; (void)ws_size;
  const float* x = (const float*)d_in[0];
  const float* Wq = (const float*)d_in[1];
  const float* Wk = (const float*)d_in[2];
  const float* Wv = (const float*)d_in[3];
  const float* Wo = (const float*)d_in[4];

  const size_t SZ = (size_t)8192 * 1024;
  uint16_t* ws = (uint16_t*)d_ws;
  uint16_t* qraw = ws;
  uint16_t* kraw = ws + SZ;
  uint16_t* vraw = ws + 2 * SZ;
  uint16_t* Qh = ws + 3 * SZ;
  uint16_t* Kh = ws + 4 * SZ;
  uint16_t* Vh = ws + 5 * SZ;
  uint16_t* Oat = qraw;  // reuse: qraw dead after rope_pack

  dim3 gg(8192 / 128, 1024 / 128);
  gemm_bf16_wmma<true, false><<<gg, 256, 0, stream>>>(x, Wq, qraw, 8192, 1024, 1024);
  gemm_bf16_wmma<true, false><<<gg, 256, 0, stream>>>(x, Wk, kraw, 8192, 1024, 1024);
  gemm_bf16_wmma<true, false><<<gg, 256, 0, stream>>>(x, Wv, vraw, 8192, 1024, 1024);

  rope_pack<<<(4 * 2048 * 16 * 32) / 256, 256, 0, stream>>>(qraw, kraw, vraw,
                                                            Qh, Kh, Vh);

  attn_flash<<<dim3(16, 4 * 16), 256, 0, stream>>>(Qh, Kh, Vh, Oat);

  gemm_bf16_wmma<false, true><<<gg, 256, 0, stream>>>(Oat, Wo, (float*)d_out,
                                                      8192, 1024, 1024);
}
